// DecoderRNN_90787018703706
// MI455X (gfx1250) — matrix operand order repaired and checked by reading, not compiled, not run on hardware
//
#include <hip/hip_runtime.h>

// ---------------------------------------------------------------- constants
#define EMBD   512
#define VOCAB  32000
#define ATTN   512
#define ENC_H  2048
#define DEC_H  1024
#define BATCH  64
#define NFEAT  196
#define TLEN   32
#define STEPS  31          // T-1
#define XDIM   (EMBD + ENC_H)   // 2560
#define G4     (4 * DEC_H)      // 4096

typedef unsigned short u16;
typedef __attribute__((ext_vector_type(16))) __bf16 v16bf;
typedef __attribute__((ext_vector_type(8)))  float  v8f;
typedef __attribute__((ext_vector_type(4)))  int    v4i;

#if __has_builtin(__builtin_amdgcn_global_load_async_to_lds_b128) && \
    __has_builtin(__builtin_amdgcn_s_wait_asynccnt)
#define HAVE_ASYNC_LDS 1
#else
#define HAVE_ASYNC_LDS 0
#endif

// ---------------------------------------------------------------- helpers
__device__ __forceinline__ u16 f2b(float f) {            // f32 -> bf16 (RNE)
  unsigned u = __builtin_bit_cast(unsigned, f);
  return (u16)((u + 0x7FFFu + ((u >> 16) & 1u)) >> 16);
}
__device__ __forceinline__ float b2f(u16 s) {
  unsigned u = ((unsigned)s) << 16;
  return __builtin_bit_cast(float, u);
}
__device__ __forceinline__ float sigm(float x) { return 1.f / (1.f + expf(-x)); }

union Frag16 { unsigned int u[8]; v16bf v; };
union Q16    { uint4 q; unsigned int u[4]; };

// =================================================================
// Generic bf16 WMMA GEMM:  C[M,N] (+)= A[M,K] @ B[K,N] (+ bias)
// Row-major A and B. Block = 128 threads (4 waves) -> 64x64 C tile.
// Each wave: 16 rows x 64 cols = four 16x16 WMMA accumulators.
// Double-buffered LDS staging; A tile uses the gfx1250 async
// global->LDS copy (ASYNCcnt) when available, B tile is transposed
// at 2-byte granularity through VGPRs. global_prefetch_b8 warms the
// next-next K tile (weights are L2-resident after step 0).
// M % 64 == 0, N % 64 == 0, K % 32 == 0 (all shapes here satisfy this).
// =================================================================
template <bool ACC, bool BF16OUT>
__global__ __launch_bounds__(128)
void gemm_bf16_wmma(const u16* __restrict__ A, int lda,
                    const u16* __restrict__ B, int ldb,
                    float* __restrict__ C, u16* __restrict__ Cb, long long ldc,
                    const float* __restrict__ bias, int K) {
  __shared__ u16 lds_a[2][64][36];   // [buf][m][k] (+pad)
  __shared__ u16 lds_b[2][64][36];   // [buf][n][k] transposed (+pad)

  const int tid  = threadIdx.x;
  const int wave = tid >> 5;
  const int lane = tid & 31;
  const int row  = lane & 15;     // M (A frag) / N (B frag) index within tile
  const int hi   = lane >> 4;     // lane-half selector
  const int rowBase = wave * 16;

  const long long blockN = (long long)blockIdx.x * 64;
  const long long blockM = (long long)blockIdx.y * 64;

  auto stageA = [&](int buf, int k0) {
#pragma unroll
    for (int j = 0; j < 2; ++j) {
      int chunk = tid + 128 * j;          // 0..255
      int r  = chunk >> 2;
      int cc = (chunk & 3) * 8;
      const u16* gsrc = A + (size_t)(blockM + r) * lda + k0 + cc;
      u16* ldst = &lds_a[buf][r][cc];
#if HAVE_ASYNC_LDS
      __builtin_amdgcn_global_load_async_to_lds_b128(
          (__attribute__((address_space(1))) v4i*)gsrc,
          (__attribute__((address_space(3))) v4i*)ldst, 0, 0);
#else
      Q16 g;
      g.q = *(const uint4*)gsrc;
      *(unsigned int*)(ldst + 0) = g.u[0];
      *(unsigned int*)(ldst + 2) = g.u[1];
      *(unsigned int*)(ldst + 4) = g.u[2];
      *(unsigned int*)(ldst + 6) = g.u[3];
#endif
    }
  };
  auto stageB = [&](int buf, int k0) {
#pragma unroll
    for (int j = 0; j < 8; ++j) {
      int idx = tid + 128 * j;            // 0..1023
      int k = idx >> 5;
      int n = (idx & 31) * 2;
      unsigned int g = *(const unsigned int*)(B + (size_t)(k0 + k) * ldb + blockN + n);
      lds_b[buf][n][k]     = (u16)(g & 0xFFFFu);
      lds_b[buf][n + 1][k] = (u16)(g >> 16);
    }
  };

  v8f acc[4];
#pragma unroll
  for (int ct = 0; ct < 4; ++ct) {
#pragma unroll
    for (int j = 0; j < 8; ++j) {
      if (ACC) {
        long long gm = blockM + rowBase + j + 8 * hi;
        long long gc = blockN + ct * 16 + row;
        acc[ct][j] = C[gm * ldc + gc];
      } else {
        acc[ct][j] = 0.f;
      }
    }
  }

  const int nit = K >> 5;
  stageA(0, 0);
  stageB(0, 0);
#if HAVE_ASYNC_LDS
  __builtin_amdgcn_s_wait_asynccnt(0);
#endif
  __syncthreads();

  for (int it = 0; it < nit; ++it) {
    const int k0  = it << 5;
    const int buf = it & 1;
    // stage next tile into the other buffer while computing from this one
    if (it + 1 < nit) {
      stageA(buf ^ 1, k0 + 32);
      stageB(buf ^ 1, k0 + 32);
    }
    // prefetch two tiles ahead (global_prefetch_b8): warm WGP$/L2 path
    if (k0 + 64 < K) {
      __builtin_prefetch(A + (size_t)(blockM + (tid >> 1)) * lda + k0 + 64, 0, 0);
      __builtin_prefetch(B + (size_t)(k0 + 64 + (tid >> 5) * 8) * ldb + blockN, 0, 0);
    }

    // ---- A fragment (ISA 16-bit A 16x32 layout: lane=M, K split by lane half)
    Frag16 af;
#pragma unroll
    for (int i = 0; i < 8; ++i) {
      int k = (i < 4) ? (hi * 8 + 2 * i) : (16 + hi * 8 + 2 * (i - 4));
      af.u[i] = *(const unsigned int*)&lds_a[buf][rowBase + row][k];
    }
    // ---- 4 column tiles
#pragma unroll
    for (int ct = 0; ct < 4; ++ct) {
      Frag16 bfr;
      int col = ct * 16 + row;
#pragma unroll
      for (int i = 0; i < 8; ++i) {
        int k = (i < 4) ? (hi * 8 + 2 * i) : (16 + hi * 8 + 2 * (i - 4));
        bfr.u[i] = *(const unsigned int*)&lds_b[buf][col][k];
      }
      acc[ct] = __builtin_amdgcn_wmma_f32_16x16x32_bf16(
          false, af.v, false, bfr.v, (short)0, acc[ct], false, false);
    }
#if HAVE_ASYNC_LDS
    __builtin_amdgcn_s_wait_asynccnt(0);
#endif
    __syncthreads();
  }

  // ---- epilogue: bias + store (C layout: m = vgpr + 8*hi, n = lane&15)
#pragma unroll
  for (int ct = 0; ct < 4; ++ct) {
    long long gc = blockN + ct * 16 + row;
    float bv = bias ? bias[gc] : 0.f;
#pragma unroll
    for (int j = 0; j < 8; ++j) {
      long long gm = blockM + rowBase + j + 8 * hi;
      float v = acc[ct][j] + bv;
      if (BF16OUT) Cb[gm * ldc + gc] = f2b(v);
      else         C[gm * ldc + gc]  = v;
    }
  }
}

// ================================================================= utility
__global__ void f32_to_bf16_k(const float* __restrict__ in, u16* __restrict__ out, size_t n) {
  size_t i = (size_t)blockIdx.x * blockDim.x + threadIdx.x;
  size_t stride = (size_t)gridDim.x * blockDim.x;
  for (; i < n; i += stride) out[i] = f2b(in[i]);
}

__global__ void bias_sum_k(const float* a, const float* b, float* o, int n) {
  int i = blockIdx.x * blockDim.x + threadIdx.x;
  if (i < n) o[i] = a[i] + b[i];
}

// embedding gather for captions[:, :STEPS] -> bf16 [B, STEPS, EMBD]
__global__ void embed_k(const float* __restrict__ tbl, const int* __restrict__ cap,
                        u16* __restrict__ embB) {
  size_t idx = (size_t)blockIdx.x * blockDim.x + threadIdx.x;
  if (idx >= (size_t)BATCH * STEPS * EMBD) return;
  int e  = idx & (EMBD - 1);
  int bt = idx >> 9;
  int t  = bt % STEPS;
  int b  = bt / STEPS;
  int tok = cap[b * TLEN + t];
  embB[idx] = f2b(tbl[(size_t)tok * EMBD + e]);
}

__global__ void feat_mean_k(const float* __restrict__ features, float* __restrict__ fm) {
  size_t idx = (size_t)blockIdx.x * blockDim.x + threadIdx.x;
  if (idx >= (size_t)BATCH * ENC_H) return;
  int b = idx >> 11, k = idx & (ENC_H - 1);
  float s = 0.f;
  for (int n = 0; n < NFEAT; ++n)
    s += features[((size_t)b * NFEAT + n) * ENC_H + k];
  fm[idx] = s * (1.f / (float)NFEAT);
}

// h0 = fm@W_h0 + b_h0 ; c0 = fm@W_c0 + b_c0  (one-time, f32 VALU)
__global__ void init_hc_k(const float* __restrict__ fm,
                          const float* __restrict__ Wh0, const float* __restrict__ bh0,
                          const float* __restrict__ Wc0, const float* __restrict__ bc0,
                          float* __restrict__ h, float* __restrict__ c, u16* __restrict__ hB) {
  int idx = blockIdx.x * blockDim.x + threadIdx.x;        // 65536
  if (idx >= BATCH * DEC_H) return;
  int b = idx >> 10, j = idx & (DEC_H - 1);
  const float* f = fm + (size_t)b * ENC_H;
  float ah = bh0[j], ac = bc0[j];
  for (int k = 0; k < ENC_H; ++k) {
    float fv = f[k];
    ah += fv * Wh0[(size_t)k * DEC_H + j];
    ac += fv * Wc0[(size_t)k * DEC_H + j];
  }
  h[idx] = ah; c[idx] = ac; hB[idx] = f2b(ah);
}

// scores[b,n] = sum_a tanh(Uf[b,n,a] + Wh[b,a]) * W_a[a] + b_a   (1 wave/row)
__global__ __launch_bounds__(256)
void attn_scores_k(const u16* __restrict__ Uf, const float* __restrict__ Wh,
                   const float* __restrict__ W_a, const float* __restrict__ b_a,
                   float* __restrict__ scores) {
  int gw = (blockIdx.x * blockDim.x + threadIdx.x) >> 5;
  int lane = threadIdx.x & 31;
  if (gw >= BATCH * NFEAT) return;
  int b = gw / NFEAT;
  const u16* u = Uf + (size_t)gw * ATTN;
  const float* w = Wh + (size_t)b * ATTN;
  float s = 0.f;
  for (int a = lane; a < ATTN; a += 32)
    s += tanhf(b2f(u[a]) + w[a]) * W_a[a];
  for (int off = 16; off; off >>= 1) s += __shfl_xor(s, off, 32);
  if (lane == 0) scores[gw] = s + b_a[0];
}

// softmax over N, write alpha, context = alpha@features, x = [emb_t, context]
__global__ __launch_bounds__(256)
void softmax_ctx_k(const float* __restrict__ scores, const u16* __restrict__ fB,
                   const u16* __restrict__ embB, float* __restrict__ attn_out,
                   u16* __restrict__ xB, int t) {
  int b = blockIdx.x, tid = threadIdx.x;
  __shared__ float red[256];
  __shared__ float sA[NFEAT];
  float v = (tid < NFEAT) ? scores[b * NFEAT + tid] : -1e30f;
  red[tid] = v; __syncthreads();
  for (int s = 128; s; s >>= 1) { if (tid < s) red[tid] = fmaxf(red[tid], red[tid + s]); __syncthreads(); }
  float m = red[0]; __syncthreads();
  float e = (tid < NFEAT) ? expf(v - m) : 0.f;
  red[tid] = e; __syncthreads();
  for (int s = 128; s; s >>= 1) { if (tid < s) red[tid] += red[tid + s]; __syncthreads(); }
  float inv = 1.f / red[0];
  if (tid < NFEAT) {
    float a = e * inv;
    sA[tid] = a;
    attn_out[((size_t)b * STEPS + t) * NFEAT + tid] = a;
  }
  __syncthreads();
  // x[:, 0:EMBD] = emb_t
  for (int j = tid; j < EMBD; j += 256)
    xB[(size_t)b * XDIM + j] = embB[((size_t)b * STEPS + t) * EMBD + j];
  // x[:, EMBD:] = context
  for (int j = tid; j < ENC_H; j += 256) {
    float acc = 0.f;
    for (int n = 0; n < NFEAT; ++n)
      acc += sA[n] * b2f(fB[((size_t)b * NFEAT + n) * ENC_H + j]);
    xB[(size_t)b * XDIM + EMBD + j] = f2b(acc);
  }
}

// LSTM pointwise:  gates already hold x@W_ih + h@W_hh + b_ih + b_hh
__global__ void lstm_step_k(const float* __restrict__ gates,
                            float* __restrict__ h, float* __restrict__ c,
                            u16* __restrict__ hB) {
  int idx = blockIdx.x * blockDim.x + threadIdx.x;        // 65536
  if (idx >= BATCH * DEC_H) return;
  int b = idx >> 10, j = idx & (DEC_H - 1);
  const float* g = gates + (size_t)b * G4;
  float i_ = sigm(g[j]);
  float f_ = sigm(g[DEC_H + j]);
  float gg = tanhf(g[2 * DEC_H + j]);
  float o_ = sigm(g[3 * DEC_H + j]);
  float cn = f_ * c[idx] + i_ * gg;
  float hn = o_ * tanhf(cn);
  c[idx] = cn; h[idx] = hn; hB[idx] = f2b(hn);
}

// ================================================================= host
extern "C" void kernel_launch(void* const* d_in, const int* in_sizes, int n_in,
                              void* d_out, int out_size, void* d_ws, size_t ws_size,
                              hipStream_t stream) {
  const float* features = (const float*)d_in[0];
  const int*   captions = (const int*)d_in[1];
  const float* emb_tbl  = (const float*)d_in[2];
  const float* W_u  = (const float*)d_in[3];
  const float* b_u  = (const float*)d_in[4];
  const float* W_w  = (const float*)d_in[5];
  const float* b_w  = (const float*)d_in[6];
  const float* W_a  = (const float*)d_in[7];
  const float* b_a  = (const float*)d_in[8];
  const float* W_ih = (const float*)d_in[9];
  const float* b_ih = (const float*)d_in[10];
  const float* W_hh = (const float*)d_in[11];
  const float* b_hh = (const float*)d_in[12];
  const float* W_fc = (const float*)d_in[13];
  const float* b_fc = (const float*)d_in[14];
  const float* W_h0 = (const float*)d_in[15];
  const float* b_h0 = (const float*)d_in[16];
  const float* W_c0 = (const float*)d_in[17];
  const float* b_c0 = (const float*)d_in[18];

  float* out = (float*)d_out;
  float* preds_out = out;                                        // [B, STEPS, VOCAB]
  float* attn_out  = out + (size_t)BATCH * STEPS * VOCAB;        // [B, STEPS, NFEAT]

  // ---- workspace layout (256B aligned)
  char* ws = (char*)d_ws;
  size_t off = 0;
#define WS(T, name, count) \
  T* name = (T*)(ws + off); off += (((size_t)(count) * sizeof(T)) + 255) & ~(size_t)255;
  WS(u16,   fB,   (size_t)BATCH * NFEAT * ENC_H)   // features bf16
  WS(u16,   WuB,  (size_t)ENC_H * ATTN)
  WS(u16,   WwB,  (size_t)DEC_H * ATTN)
  WS(u16,   WihB, (size_t)XDIM * G4)
  WS(u16,   WhhB, (size_t)DEC_H * G4)
  WS(u16,   WfcB, (size_t)DEC_H * VOCAB)
  WS(u16,   UfB,  (size_t)BATCH * NFEAT * ATTN)    // bf16 Uf
  WS(u16,   embB, (size_t)BATCH * STEPS * EMBD)
  WS(float, fm,   (size_t)BATCH * ENC_H)
  WS(float, hbuf, (size_t)BATCH * DEC_H)
  WS(float, cbuf, (size_t)BATCH * DEC_H)
  WS(u16,   hB,   (size_t)BATCH * DEC_H)
  WS(u16,   xB,   (size_t)BATCH * XDIM)
  WS(float, Wh,   (size_t)BATCH * ATTN)
  WS(float, scores, (size_t)BATCH * NFEAT)
  WS(float, gates,  (size_t)BATCH * G4)
  WS(float, bsum,   (size_t)G4)
#undef WS

  auto cvt = [&](const float* src, u16* dst, size_t n) {
    int blocks = (int)((n + 2047) / 2048); if (blocks > 8192) blocks = 8192;
    f32_to_bf16_k<<<blocks, 256, 0, stream>>>(src, dst, n);
  };

  // ---- one-time precompute
  cvt(features, fB, (size_t)BATCH * NFEAT * ENC_H);
  cvt(W_u,  WuB,  (size_t)ENC_H * ATTN);
  cvt(W_w,  WwB,  (size_t)DEC_H * ATTN);
  cvt(W_ih, WihB, (size_t)XDIM * G4);
  cvt(W_hh, WhhB, (size_t)DEC_H * G4);
  cvt(W_fc, WfcB, (size_t)DEC_H * VOCAB);
  bias_sum_k<<<G4 / 256, 256, 0, stream>>>(b_ih, b_hh, bsum, G4);
  embed_k<<<((size_t)BATCH * STEPS * EMBD + 255) / 256, 256, 0, stream>>>(emb_tbl, captions, embB);
  feat_mean_k<<<(BATCH * ENC_H + 255) / 256, 256, 0, stream>>>(features, fm);
  init_hc_k<<<(BATCH * DEC_H + 255) / 256, 256, 0, stream>>>(fm, W_h0, b_h0, W_c0, b_c0,
                                                             hbuf, cbuf, hB);
  // Uf = features @ W_u + b_u  -> bf16  [12544 x 512], K=2048
  gemm_bf16_wmma<false, true><<<dim3(ATTN / 64, (BATCH * NFEAT) / 64), 128, 0, stream>>>(
      fB, ENC_H, WuB, ATTN, nullptr, UfB, (long long)ATTN, b_u, ENC_H);

  // ---- 31 recurrent steps
  for (int t = 0; t < STEPS; ++t) {
    // Wh = h @ W_w + b_w   [64 x 512], K=1024
    gemm_bf16_wmma<false, false><<<dim3(ATTN / 64, 1), 128, 0, stream>>>(
        hB, DEC_H, WwB, ATTN, Wh, nullptr, (long long)ATTN, b_w, DEC_H);
    // scores
    attn_scores_k<<<(BATCH * NFEAT * 32) / 256, 256, 0, stream>>>(UfB, Wh, W_a, b_a, scores);
    // softmax + context + x assembly (+ alpha output)
    softmax_ctx_k<<<BATCH, 256, 0, stream>>>(scores, fB, embB, attn_out, xB, t);
    // gates = x @ W_ih + (b_ih + b_hh)   [64 x 4096], K=2560
    gemm_bf16_wmma<false, false><<<dim3(G4 / 64, 1), 128, 0, stream>>>(
        xB, XDIM, WihB, G4, gates, nullptr, (long long)G4, bsum, XDIM);
    // gates += h @ W_hh                  [64 x 4096], K=1024
    gemm_bf16_wmma<true, false><<<dim3(G4 / 64, 1), 128, 0, stream>>>(
        hB, DEC_H, WhhB, G4, gates, nullptr, (long long)G4, nullptr, DEC_H);
    // LSTM pointwise -> h, c, hB
    lstm_step_k<<<(BATCH * DEC_H + 255) / 256, 256, 0, stream>>>(gates, hbuf, cbuf, hB);
    // logits = h_new @ W_fc + b_fc -> preds[:, t, :]   [64 x 32000], K=1024
    gemm_bf16_wmma<false, false><<<dim3(VOCAB / 64, 1), 128, 0, stream>>>(
        hB, DEC_H, WfcB, VOCAB, preds_out + (size_t)t * VOCAB, nullptr,
        (long long)STEPS * VOCAB, b_fc, DEC_H);
  }
}
